// RNNCommunicatingAgent_20040317403416
// MI455X (gfx1250) — matrix operand order repaired and checked by reading, not compiled, not run on hardware
//
#include <hip/hip_runtime.h>
#include <cstddef>
#include <cstdint>

// ---------------------------------------------------------------------------
// MI455X (gfx1250) implementation.
// Strategy: bandwidth-bound problem (64 GFLOP vs ~119 MB bf16 weights).
// Convert f32 weights -> bf16 once per launch into d_ws, run all large GEMMs
// through v_wmma_f32_16x16x32_bf16 (wave32, 16x64 tile per wave), keep all
// activations/accumulation in f32.
// ---------------------------------------------------------------------------

typedef __attribute__((ext_vector_type(16))) __bf16 v16bf;
typedef __attribute__((ext_vector_type(8)))  __bf16 v8bf;
typedef __attribute__((ext_vector_type(8)))  float  v8f;

#define Bq 256
#define OBSq 512
#define NAq 64
#define MSGq 2048
#define Hq 1024
#define INW 3136   // 2*OBS + NA + MSG

__device__ __forceinline__ unsigned short f2bf(float f) {
  unsigned u = __float_as_uint(f);
  unsigned r = u + 0x7FFFu + ((u >> 16) & 1u);   // round-to-nearest-even
  return (unsigned short)(r >> 16);
}
__device__ __forceinline__ float sigmoidf_(float x) { return 1.0f / (1.0f + __expf(-x)); }

// ---------------------------------------------------------------------------
// f32 -> bf16 bulk convert (weights)
// ---------------------------------------------------------------------------
__global__ void cvt_bf16_kernel(unsigned short* __restrict__ dst,
                                const float* __restrict__ src, int n) {
  for (int i = blockIdx.x * blockDim.x + threadIdx.x; i < n;
       i += gridDim.x * blockDim.x)
    dst[i] = f2bf(src[i]);
}

// submatrix copy f32 -> bf16 (activation packing / concat)
__global__ void copy_sub_bf16_kernel(unsigned short* __restrict__ dst, int ldd, int dof,
                                     const float* __restrict__ src, int lds, int sof,
                                     int rows, int cols) {
  int total = rows * cols;
  for (int i = blockIdx.x * blockDim.x + threadIdx.x; i < total;
       i += gridDim.x * blockDim.x) {
    int r = i / cols, c = i - r * cols;
    dst[(size_t)r * ldd + dof + c] = f2bf(src[(size_t)r * lds + sof + c]);
  }
}

// submatrix copy f32 -> f32
__global__ void copy_sub_f32_kernel(float* __restrict__ dst, int ldd, int dof,
                                    const float* __restrict__ src, int lds, int sof,
                                    int rows, int cols) {
  int total = rows * cols;
  for (int i = blockIdx.x * blockDim.x + threadIdx.x; i < total;
       i += gridDim.x * blockDim.x) {
    int r = i / cols, c = i - r * cols;
    dst[(size_t)r * ldd + dof + c] = src[(size_t)r * lds + sof + c];
  }
}

// broadcast one f32 row vector to every row, converting to bf16
__global__ void bcast_rows_bf16_kernel(unsigned short* __restrict__ dst, int ldd, int dof,
                                       const float* __restrict__ src, int rows, int cols) {
  int total = rows * cols;
  for (int i = blockIdx.x * blockDim.x + threadIdx.x; i < total;
       i += gridDim.x * blockDim.x) {
    int r = i / cols, c = i - r * cols;
    dst[(size_t)r * ldd + dof + c] = f2bf(src[c]);
  }
}

// ---------------------------------------------------------------------------
// GEMM:  C[M,N] = act( A[M,K](bf16) * W[N,K]^T(bf16) + bias )   f32 accumulate
// wave32; one wave -> 16(M) x 64(N) tile; K stepped by 32 via
// v_wmma_f32_16x16x32_bf16. Requires M%16==0, N%64==0, K%32==0.
// ---------------------------------------------------------------------------
__global__ __launch_bounds__(128) void gemm_bf16_kernel(
    const unsigned short* __restrict__ A, int lda,
    const unsigned short* __restrict__ W,   // [N,K] row-major
    const float* __restrict__ bias,
    float* __restrict__ C, int ldc,
    int M, int N, int K, int relu) {
  int wid  = blockIdx.x * (blockDim.x >> 5) + (threadIdx.x >> 5);
  int lane = threadIdx.x & 31;
  int tilesM = M >> 4, tilesN = N >> 6;
  if (wid >= tilesM * tilesN) return;           // wave-uniform: EXEC stays all-1s
  int tm = wid % tilesM, tn = wid / tilesM;     // waves in a block share N-tile weights
  int m0 = tm << 4, n0 = tn << 6;
  int l = lane & 15, hi = lane >> 4;

  // A fragment addressing (this code's contiguous k order, same as B):
  //   lanes 0-15:  elems 0-15 = K k0..k0+15
  //   lanes 16-31: elems 0-15 = K k0+16..k0+31
  const unsigned short* Arow = A + (size_t)(m0 + l) * lda + hi * 16;
  // B fragment addressing (K x 16 tile): lane n holds column n, 16 contiguous K
  const unsigned short* W0 = W + (size_t)(n0 + l) * K + hi * 16;
  const size_t wstep = (size_t)16 * K;

  v8f acc[4] = {v8f{}, v8f{}, v8f{}, v8f{}};

  for (int k0 = 0; k0 < K; k0 += 32) {
    v8bf alo = *(const v8bf*)(Arow + k0);
    v8bf ahi = *(const v8bf*)(Arow + k0 + 8);
    v16bf a;
#pragma unroll
    for (int i = 0; i < 8; ++i) { a[i] = alo[i]; a[i + 8] = ahi[i]; }

    v16bf b0 = *(const v16bf*)(W0 + k0);
    v16bf b1 = *(const v16bf*)(W0 + wstep + k0);
    v16bf b2 = *(const v16bf*)(W0 + 2 * wstep + k0);
    v16bf b3 = *(const v16bf*)(W0 + 3 * wstep + k0);

    if (k0 + 128 < K) {                       // stream-ahead: global_prefetch_b8
      __builtin_prefetch(W0 + k0 + 128);
      __builtin_prefetch(W0 + wstep + k0 + 128);
      __builtin_prefetch(W0 + 2 * wstep + k0 + 128);
      __builtin_prefetch(W0 + 3 * wstep + k0 + 128);
    }

    acc[0] = __builtin_amdgcn_wmma_f32_16x16x32_bf16(false, a, false, b0, (short)0, acc[0], false, false);
    acc[1] = __builtin_amdgcn_wmma_f32_16x16x32_bf16(false, a, false, b1, (short)0, acc[1], false, false);
    acc[2] = __builtin_amdgcn_wmma_f32_16x16x32_bf16(false, a, false, b2, (short)0, acc[2], false, false);
    acc[3] = __builtin_amdgcn_wmma_f32_16x16x32_bf16(false, a, false, b3, (short)0, acc[3], false, false);
  }

  // C/D layout: VGPR r -> row m0 + r + hi*8 ; lane&15 -> column within tile
#pragma unroll
  for (int j = 0; j < 4; ++j) {
    int col = n0 + j * 16 + l;
    float bs = bias ? bias[col] : 0.0f;
#pragma unroll
    for (int r = 0; r < 8; ++r) {
      int row = m0 + hi * 8 + r;
      float v = acc[j][r] + bs;
      if (relu) v = fmaxf(v, 0.0f);
      C[(size_t)row * ldc + col] = v;
    }
  }
}

// ---------------------------------------------------------------------------
// GRU elementwise combine: h' = (1-z)*n + z*h
// gi,gh: [256,3072]; h read at (ldh,offh); dst: [256,1024]
// ---------------------------------------------------------------------------
__global__ void gru_combine_kernel(const float* __restrict__ gi,
                                   const float* __restrict__ gh,
                                   const float* __restrict__ h, int ldh, int offh,
                                   float* __restrict__ dst) {
  int total = Bq * Hq;
  for (int i = blockIdx.x * blockDim.x + threadIdx.x; i < total;
       i += gridDim.x * blockDim.x) {
    int b = i >> 10, j = i & 1023;
    size_t o = (size_t)b * 3072 + j;
    float r  = sigmoidf_(gi[o] + gh[o]);
    float z  = sigmoidf_(gi[o + 1024] + gh[o + 1024]);
    float nn = tanhf(gi[o + 2048] + r * gh[o + 2048]);
    dst[i] = (1.0f - z) * nn + z * h[(size_t)b * ldh + offh + j];
  }
}

// softmax over rows of 64 (wave32: 2 elems/lane, shfl_xor reduction)
__global__ void softmax64_kernel(float* __restrict__ q, int rows) {
  int wid  = (blockIdx.x * blockDim.x + threadIdx.x) >> 5;
  int lane = threadIdx.x & 31;
  if (wid >= rows) return;
  float* r = q + (size_t)wid * 64;
  float a = r[lane], b = r[lane + 32];
  float m = fmaxf(a, b);
  for (int o = 16; o; o >>= 1) m = fmaxf(m, __shfl_xor(m, o, 32));
  a = __expf(a - m); b = __expf(b - m);
  float s = a + b;
  for (int o = 16; o; o >>= 1) s += __shfl_xor(s, o, 32);
  float inv = 1.0f / s;
  r[lane] = a * inv; r[lane + 32] = b * inv;
}

// fa = transpose(qs,(1,0,2)).reshape(B,16320): gather straight into bf16 concat buf
__global__ void gather_fa_kernel(unsigned short* __restrict__ dst /* ld 16896 */,
                                 const float* __restrict__ q /* [256,16320] */) {
  const int total = Bq * 16320;
  for (int f = blockIdx.x * blockDim.x + threadIdx.x; f < total;
       f += gridDim.x * blockDim.x) {
    int b = f / 16320, j = f - b * 16320;
    int t = f >> 14;             // f / (256*64)
    int bb = (f >> 6) & 255;
    int n = f & 63;
    dst[(size_t)b * 16896 + j] = f2bf(q[(size_t)bb * 16320 + t * 64 + n]);
  }
}

// vv[t,b,v] = am_vs(taus[t,b]) ; taus = concat(obs_t, act_t)
__global__ void vv_kernel(const float* __restrict__ obs3, const float* __restrict__ act3,
                          const float* __restrict__ vw, const float* __restrict__ vb,
                          float* __restrict__ vv) {
  int tb = blockIdx.x;                 // 0..767
  int v = threadIdx.x;
  if (v >= 8) return;
  int t = tb >> 8, b = tb & 255;
  const float* o = obs3 + ((size_t)t * Bq + b) * OBSq;
  const float* a = act3 + ((size_t)t * Bq + b) * NAq;
  float s = vb[v];
  for (int i = 0; i < OBSq; ++i) s += o[i] * vw[v * 576 + i];
  for (int i = 0; i < NAq;  ++i) s += a[i] * vw[v * 576 + OBSq + i];
  vv[(size_t)tb * 8 + v] = s;
}

// alpha[b,t] = softmax_t( dot(qv[b], kk[t,b]) / 32 )
__global__ void attn_alpha_kernel(const float* __restrict__ qv,
                                  const float* __restrict__ kk,
                                  float* __restrict__ alpha) {
  int b = (blockIdx.x * blockDim.x + threadIdx.x) >> 5;
  int lane = threadIdx.x & 31;
  if (b >= Bq) return;
  float s[3];
#pragma unroll
  for (int t = 0; t < 3; ++t) {
    float p = 0.0f;
    for (int h = lane; h < Hq; h += 32)
      p += qv[(size_t)b * Hq + h] * kk[((size_t)t * Bq + b) * Hq + h];
    for (int o = 16; o; o >>= 1) p += __shfl_xor(p, o, 32);
    s[t] = p * (1.0f / 32.0f);   // 1/sqrt(1024)
  }
  float m = fmaxf(s[0], fmaxf(s[1], s[2]));
  float e0 = __expf(s[0] - m), e1 = __expf(s[1] - m), e2 = __expf(s[2] - m);
  float inv = 1.0f / (e0 + e1 + e2);
  if (lane == 0) {
    alpha[b * 3 + 0] = e0 * inv;
    alpha[b * 3 + 1] = e1 * inv;
    alpha[b * 3 + 2] = e2 * inv;
  }
}

// new_message[b,v] = sum_t alpha[b,t] * vv[t,b,v]
__global__ void new_msg_kernel(const float* __restrict__ alpha,
                               const float* __restrict__ vv,
                               float* __restrict__ nm) {
  int idx = blockIdx.x * blockDim.x + threadIdx.x;
  if (idx >= Bq * 8) return;
  int b = idx >> 3, v = idx & 7;
  float s = 0.0f;
#pragma unroll
  for (int t = 0; t < 3; ++t)
    s += alpha[b * 3 + t] * vv[((size_t)t * Bq + b) * 8 + v];
  nm[idx] = s;
}

// ---------------------------------------------------------------------------
// host side
// ---------------------------------------------------------------------------
static inline int cdiv_(int a, int b) { return (a + b - 1) / b; }
static inline int imin_(int a, int b) { return a < b ? a : b; }

extern "C" void kernel_launch(void* const* d_in, const int* in_sizes, int n_in,
                              void* d_out, int out_size, void* d_ws, size_t ws_size,
                              hipStream_t stream) {
  (void)in_sizes; (void)n_in; (void)out_size; (void)ws_size;
  const float* inp     = (const float*)d_in[0];
  const float* hidden  = (const float*)d_in[1];
  const float* pi_fc1_w = (const float*)d_in[2];  const float* pi_fc1_b = (const float*)d_in[3];
  const float* pi_fc2_w = (const float*)d_in[4];  const float* pi_fc2_b = (const float*)d_in[5];
  const float* fa_fc1_w = (const float*)d_in[6];  const float* fa_fc1_b = (const float*)d_in[7];
  const float* fa_fc2_w = (const float*)d_in[8];  const float* fa_fc2_b = (const float*)d_in[9];
  const float* fo_fc1_w = (const float*)d_in[10]; const float* fo_fc1_b = (const float*)d_in[11];
  const float* fo_fc2_w = (const float*)d_in[12]; const float* fo_fc2_b = (const float*)d_in[13];
  const float* am_qs_w  = (const float*)d_in[14]; const float* am_qs_b  = (const float*)d_in[15];
  const float* am_ks_w  = (const float*)d_in[16]; const float* am_ks_b  = (const float*)d_in[17];
  const float* am_vs_w  = (const float*)d_in[18]; const float* am_vs_b  = (const float*)d_in[19];
  const float* pi_wih = (const float*)d_in[20]; const float* pi_bih = (const float*)d_in[21];
  const float* pi_whh = (const float*)d_in[22]; const float* pi_bhh = (const float*)d_in[23];
  const float* fa_wih = (const float*)d_in[24]; const float* fa_bih = (const float*)d_in[25];
  const float* fa_whh = (const float*)d_in[26]; const float* fa_bhh = (const float*)d_in[27];
  const float* fo_wih = (const float*)d_in[28]; const float* fo_bih = (const float*)d_in[29];
  const float* fo_whh = (const float*)d_in[30]; const float* fo_bhh = (const float*)d_in[31];

  // ---- workspace layout (~165 MB needed) ----
  char* base = (char*)d_ws;
  size_t off = 0;
  auto aus = [&](size_t e) { unsigned short* p = (unsigned short*)(base + off);
                             off += ((e * 2 + 255) & ~(size_t)255); return p; };
  auto af  = [&](size_t e) { float* p = (float*)(base + off);
                             off += ((e * 4 + 255) & ~(size_t)255); return p; };

  unsigned short* wb_pi_fc1 = aus(1024 * 2560);
  unsigned short* wb_pi_fc2 = aus(64 * 1024);
  unsigned short* wb_fa_fc1 = aus(1024 * 512);
  unsigned short* wb_fa_fc2 = aus((size_t)16320 * 1024);
  unsigned short* wb_fo_fc1 = aus((size_t)1024 * 16896);
  unsigned short* wb_fo_fc2 = aus(512 * 1024);
  unsigned short* wb_am_qs  = aus(1024 * 2048);
  unsigned short* wb_am_ks  = aus(1024 * 576);
  unsigned short* wb_pi_wih = aus(3072 * 1024); unsigned short* wb_pi_whh = aus(3072 * 1024);
  unsigned short* wb_fa_wih = aus(3072 * 1024); unsigned short* wb_fa_whh = aus(3072 * 1024);
  unsigned short* wb_fo_wih = aus(3072 * 1024); unsigned short* wb_fo_whh = aus(3072 * 1024);

  float* obs3 = af(3 * Bq * OBSq);
  float* act3 = af(3 * Bq * NAq);
  float* h_fa = af(Bq * Hq); float* h_fo = af(Bq * Hq);
  float* h_pi = af(Bq * Hq); float* h_pif = af(Bq * Hq);
  float* x_f32 = af(Bq * Hq);
  float* gi = af(Bq * 3072); float* gh = af(Bq * 3072);
  float* q_f32 = af((size_t)Bq * 16320);
  float* kk = af(3 * Bq * Hq);
  float* qv = af(Bq * Hq);
  float* vvb = af(3 * Bq * 8);
  float* alphab = af(Bq * 3);
  float* nmsg = af(Bq * 8);
  float* actionb = af(Bq * NAq);

  unsigned short* bf_in1 = aus((size_t)Bq * 16896);
  unsigned short* bf_x   = aus(Bq * Hq);
  unsigned short* bf_h   = aus(Bq * Hq);
  unsigned short* bf_tau = aus((size_t)3 * Bq * 576);

  // ---- helpers ----
  auto cvt = [&](unsigned short* d, const float* s, size_t n) {
    int blk = imin_(cdiv_((int)n, 256), 16384);
    cvt_bf16_kernel<<<blk, 256, 0, stream>>>(d, s, (int)n);
  };
  auto pack_bf = [&](unsigned short* d, int ldd, int dof, const float* s, int lds,
                     int sof, int rows, int cols) {
    int n = rows * cols; int blk = imin_(cdiv_(n, 256), 16384);
    copy_sub_bf16_kernel<<<blk, 256, 0, stream>>>(d, ldd, dof, s, lds, sof, rows, cols);
  };
  auto copy_f = [&](float* d, int ldd, int dof, const float* s, int lds, int sof,
                    int rows, int cols) {
    int n = rows * cols; int blk = imin_(cdiv_(n, 256), 16384);
    copy_sub_f32_kernel<<<blk, 256, 0, stream>>>(d, ldd, dof, s, lds, sof, rows, cols);
  };
  auto gemm = [&](const unsigned short* A, int lda, const unsigned short* W,
                  const float* bias, float* C, int ldc, int M, int N, int K, int relu) {
    int tiles = (M / 16) * (N / 64);
    gemm_bf16_kernel<<<cdiv_(tiles, 4), 128, 0, stream>>>(A, lda, W, bias, C, ldc, M, N, K, relu);
  };
  auto gru_step = [&](const float* x, const float* hsrc, int ldh, int offh,
                      const unsigned short* wih, const float* bih,
                      const unsigned short* whh, const float* bhh, float* hdst) {
    pack_bf(bf_x, 1024, 0, x, 1024, 0, Bq, 1024);
    pack_bf(bf_h, 1024, 0, hsrc, ldh, offh, Bq, 1024);
    gemm(bf_x, 1024, wih, bih, gi, 3072, Bq, 3072, 1024, 0);
    gemm(bf_h, 1024, whh, bhh, gh, 3072, Bq, 3072, 1024, 0);
    gru_combine_kernel<<<cdiv_(Bq * Hq, 256), 256, 0, stream>>>(gi, gh, hsrc, ldh, offh, hdst);
  };

  // ---- convert weights to bf16 (deterministic, every call) ----
  cvt(wb_pi_fc1, pi_fc1_w, (size_t)1024 * 2560);
  cvt(wb_pi_fc2, pi_fc2_w, (size_t)64 * 1024);
  cvt(wb_fa_fc1, fa_fc1_w, (size_t)1024 * 512);
  cvt(wb_fa_fc2, fa_fc2_w, (size_t)16320 * 1024);
  cvt(wb_fo_fc1, fo_fc1_w, (size_t)1024 * 16896);
  cvt(wb_fo_fc2, fo_fc2_w, (size_t)512 * 1024);
  cvt(wb_am_qs, am_qs_w, (size_t)1024 * 2048);
  cvt(wb_am_ks, am_ks_w, (size_t)1024 * 576);
  cvt(wb_pi_wih, pi_wih, (size_t)3072 * 1024); cvt(wb_pi_whh, pi_whh, (size_t)3072 * 1024);
  cvt(wb_fa_wih, fa_wih, (size_t)3072 * 1024); cvt(wb_fa_whh, fa_whh, (size_t)3072 * 1024);
  cvt(wb_fo_wih, fo_wih, (size_t)3072 * 1024); cvt(wb_fo_whh, fo_whh, (size_t)3072 * 1024);

  // ---- initial state ----
  copy_f(h_fa, 1024, 0, hidden, 3072, 0,    Bq, 1024);
  copy_f(h_fo, 1024, 0, hidden, 3072, 1024, Bq, 1024);
  copy_f(h_pi, 1024, 0, hidden, 3072, 2048, Bq, 1024);
  copy_f(obs3, OBSq, 0, inp, INW, OBSq, Bq, OBSq);        // prev_obs
  copy_f(act3, NAq,  0, inp, INW, 2 * OBSq, Bq, NAq);     // prev_act

  // ---- rollout loop (T-1 = 2 steps) ----
  for (int t = 0; t < 2; ++t) {
    const float* obs_t = obs3 + (size_t)t * Bq * OBSq;
    const float* act_t = act3 + (size_t)t * Bq * NAq;
    float* obs_n = obs3 + (size_t)(t + 1) * Bq * OBSq;
    float* act_n = act3 + (size_t)(t + 1) * Bq * NAq;

    // fa branch: x = relu(fa_fc1(o)); h_fa = GRU(x, h_fa); q = fa_fc2(h_fa)
    pack_bf(bf_in1, OBSq, 0, obs_t, OBSq, 0, Bq, OBSq);
    gemm(bf_in1, OBSq, wb_fa_fc1, fa_fc1_b, x_f32, 1024, Bq, 1024, OBSq, 1);
    gru_step(x_f32, h_fa, 1024, 0, wb_fa_wih, fa_bih, wb_fa_whh, fa_bhh, h_fa);
    pack_bf(bf_x, 1024, 0, h_fa, 1024, 0, Bq, 1024);
    gemm(bf_x, 1024, wb_fa_fc2, fa_fc2_b, q_f32, 16320, Bq, 16320, 1024, 0);
    softmax64_kernel<<<cdiv_(Bq * 255 * 32, 256), 256, 0, stream>>>(q_f32, Bq * 255);

    // fo branch input: concat(fa_shuffled(16320), a(64), o(512))
    gather_fa_kernel<<<imin_(cdiv_(Bq * 16320, 256), 16384), 256, 0, stream>>>(bf_in1, q_f32);
    pack_bf(bf_in1, 16896, 16320, act_t, NAq, 0, Bq, NAq);
    pack_bf(bf_in1, 16896, 16384, obs_t, OBSq, 0, Bq, OBSq);
    gemm(bf_in1, 16896, wb_fo_fc1, fo_fc1_b, x_f32, 1024, Bq, 1024, 16896, 1);
    gru_step(x_f32, h_fo, 1024, 0, wb_fo_wih, fo_bih, wb_fo_whh, fo_bhh, h_fo);
    pack_bf(bf_x, 1024, 0, h_fo, 1024, 0, Bq, 1024);
    gemm(bf_x, 1024, wb_fo_fc2, fo_fc2_b, obs_n, OBSq, Bq, OBSq, 1024, 0);

    // pi branch: uses current o + prev_msg + h_pi
    pack_bf(bf_in1, 2560, 0, obs_t, OBSq, 0, Bq, OBSq);
    pack_bf(bf_in1, 2560, OBSq, inp, INW, 2 * OBSq + NAq, Bq, MSGq);
    gemm(bf_in1, 2560, wb_pi_fc1, pi_fc1_b, x_f32, 1024, Bq, 1024, 2560, 1);
    gru_step(x_f32, h_pi, 1024, 0, wb_pi_wih, pi_bih, wb_pi_whh, pi_bhh, h_pi);
    pack_bf(bf_x, 1024, 0, h_pi, 1024, 0, Bq, 1024);
    gemm(bf_x, 1024, wb_pi_fc2, pi_fc2_b, act_n, NAq, Bq, NAq, 1024, 0);
  }

  // ---- attention message ----
  for (int t = 0; t < 3; ++t) {
    unsigned short* tau_t = bf_tau + (size_t)t * Bq * 576;
    pack_bf(tau_t, 576, 0, obs3 + (size_t)t * Bq * OBSq, OBSq, 0, Bq, OBSq);
    pack_bf(tau_t, 576, OBSq, act3 + (size_t)t * Bq * NAq, NAq, 0, Bq, NAq);
  }
  pack_bf(bf_in1, MSGq, 0, inp, INW, 2 * OBSq + NAq, Bq, MSGq);      // prev_msg
  gemm(bf_in1, MSGq, wb_am_qs, am_qs_b, qv, 1024, Bq, 1024, MSGq, 0);
  gemm(bf_tau, 576, wb_am_ks, am_ks_b, kk, 1024, 3 * Bq, 1024, 576, 0);
  vv_kernel<<<3 * Bq, 32, 0, stream>>>(obs3, act3, am_vs_w, am_vs_b, vvb);
  attn_alpha_kernel<<<cdiv_(Bq * 32, 128), 128, 0, stream>>>(qv, kk, alphab);
  new_msg_kernel<<<cdiv_(Bq * 8, 256), 256, 0, stream>>>(alphab, vvb, nmsg);

  // ---- final pi: pi(obs, pi_msg, h_pi0); pi_msg rows all equal flat(new_message)
  pack_bf(bf_in1, 2560, 0, inp, INW, 0, Bq, OBSq);                    // obs
  bcast_rows_bf16_kernel<<<imin_(cdiv_(Bq * MSGq, 256), 16384), 256, 0, stream>>>(
      bf_in1, 2560, OBSq, nmsg, Bq, MSGq);
  gemm(bf_in1, 2560, wb_pi_fc1, pi_fc1_b, x_f32, 1024, Bq, 1024, 2560, 1);
  gru_step(x_f32, hidden, 3072, 2048, wb_pi_wih, pi_bih, wb_pi_whh, pi_bhh, h_pif);
  pack_bf(bf_x, 1024, 0, h_pif, 1024, 0, Bq, 1024);
  gemm(bf_x, 1024, wb_pi_fc2, pi_fc2_b, actionb, NAq, Bq, NAq, 1024, 0);

  // ---- outputs: q_out[256,72] then h_out[256,3072] ----
  float* out = (float*)d_out;
  copy_f(out, 72, 0, nmsg, 8, 0, Bq, 8);
  copy_f(out, 72, 8, actionb, NAq, 0, Bq, NAq);
  float* hout = out + Bq * 72;
  copy_f(hout, 3072, 0,    h_fa,  1024, 0, Bq, 1024);
  copy_f(hout, 3072, 1024, h_fo,  1024, 0, Bq, 1024);
  copy_f(hout, 3072, 2048, h_pif, 1024, 0, Bq, 1024);
}
